// Attention_2903397892926
// MI455X (gfx1250) — matrix operand order repaired and checked by reading, not compiled, not run on hardware
//
#include <hip/hip_runtime.h>

typedef _Float16 v16h __attribute__((ext_vector_type(16)));
typedef float    v8f  __attribute__((ext_vector_type(8)));
typedef _Float16 h8   __attribute__((ext_vector_type(8)));
typedef _Float16 h4   __attribute__((ext_vector_type(4)));
typedef float    f4   __attribute__((ext_vector_type(4)));
typedef unsigned int v4u __attribute__((ext_vector_type(4)));
typedef int          v8i __attribute__((ext_vector_type(8)));
typedef int          v4i __attribute__((ext_vector_type(4)));

#define BS 4096
#define D  256
#define NS 32

// ---- WMMA fragment gathers from padded LDS tiles (merge into ds_load_b128) ----
__device__ __forceinline__ v16h load_a_frag(const _Float16* Al, int row, int stride, int hl) {
    v16h a;
#pragma unroll
    for (int v = 0; v < 8; ++v) {
        int ka = (v & 3) * 2 + (v >> 2) * 16 + hl * 8;        // A K-map (16x32 f16)
        a[2 * v]     = Al[row * stride + ka];
        a[2 * v + 1] = Al[row * stride + ka + 1];
    }
    return a;
}
__device__ __forceinline__ v16h load_b_frag(const _Float16* Bt, int col, int stride, int hl) {
    v16h b;
#pragma unroll
    for (int v = 0; v < 8; ++v) {
        int kb = (v & 3) * 2 + (v >> 2) * 16 + hl * 8;        // B K-map (32x16 f16)
        b[2 * v]     = Bt[col * stride + kb];
        b[2 * v + 1] = Bt[col * stride + kb + 1];
    }
    return b;
}

// -------------------- Phase 1a: Q projection --------------------
// Qp[b][slot][a] = sum_d q[slot][b][d] * wq[slot][d][a]; block = 128 x 256 tile
__global__ __launch_bounds__(512) void proj_q_kernel(
    const float* __restrict__ qin, const float* __restrict__ wq,
    _Float16* __restrict__ Qp)
{
    const int slot = blockIdx.y;
    const float* X = qin + (size_t)slot * BS * D;
    const float* W = wq  + (size_t)slot * D * D;
    const int b0 = blockIdx.x * 128;

    __shared__ _Float16 Al[128 * 40];   // A tile [m][k], padded
    __shared__ _Float16 Bt[256 * 40];   // B tile transposed [n][k], padded

    const int tid = threadIdx.x;
    const int lane = tid & 31;
    const int w   = tid >> 5;           // 16 waves
    const int wm  = w & 3;              // 32-row sub-tile
    const int wn  = w >> 2;             // 64-col sub-tile (of 256)
    const int hl  = lane >> 4;
    const int l16 = lane & 15;

    v8f acc[2][4];
#pragma unroll
    for (int mi = 0; mi < 2; ++mi)
#pragma unroll
        for (int ni = 0; ni < 4; ++ni) acc[mi][ni] = (v8f){};

    for (int kt = 0; kt < 8; ++kt) {
        const int k0 = kt * 32;
#pragma unroll
        for (int i = 0; i < 2; ++i) {   // A: 128x32 = 1024 f4
            int idx = tid + i * 512;
            int row = idx >> 3, c4 = idx & 7;
            f4 v = __builtin_nontemporal_load((const f4*)&X[(size_t)(b0 + row) * D + k0 + c4 * 4]);
            h4 hv = {(_Float16)v.x, (_Float16)v.y, (_Float16)v.z, (_Float16)v.w};
            *(h4*)&Al[row * 40 + c4 * 4] = hv;
        }
#pragma unroll
        for (int i = 0; i < 4; ++i) {   // B: 32x256 = 2048 f4, stored transposed
            int idx = tid + i * 512;
            int row = idx >> 6, c4 = idx & 63;
            f4 v = *(const f4*)&W[(size_t)(k0 + row) * D + c4 * 4];
            Bt[(c4 * 4 + 0) * 40 + row] = (_Float16)v.x;
            Bt[(c4 * 4 + 1) * 40 + row] = (_Float16)v.y;
            Bt[(c4 * 4 + 2) * 40 + row] = (_Float16)v.z;
            Bt[(c4 * 4 + 3) * 40 + row] = (_Float16)v.w;
        }
        __syncthreads();
        v16h a0 = load_a_frag(Al, wm * 32 + l16,      40, hl);
        v16h a1 = load_a_frag(Al, wm * 32 + 16 + l16, 40, hl);
#pragma unroll
        for (int ni = 0; ni < 4; ++ni) {
            v16h bf = load_b_frag(Bt, wn * 64 + ni * 16 + l16, 40, hl);
            acc[0][ni] = __builtin_amdgcn_wmma_f32_16x16x32_f16(false, a0, false, bf, (short)0, acc[0][ni], false, false);
            acc[1][ni] = __builtin_amdgcn_wmma_f32_16x16x32_f16(false, a1, false, bf, (short)0, acc[1][ni], false, false);
        }
        __syncthreads();
    }
#pragma unroll
    for (int mi = 0; mi < 2; ++mi)
#pragma unroll
        for (int ni = 0; ni < 4; ++ni)
#pragma unroll
            for (int r = 0; r < 8; ++r) {
                int brow = b0 + wm * 32 + mi * 16 + r + hl * 8;
                int col  = wn * 64 + ni * 16 + l16;
                Qp[(size_t)brow * (NS * D) + slot * D + col] = (_Float16)acc[mi][ni][r];
            }
}

// -------------------- Phase 1b: fused K and V projections --------------------
// Shares the staged k-tile for both weight matrices: k is read from HBM once.
__global__ __launch_bounds__(512) void proj_kv_kernel(
    const float* __restrict__ kin, const float* __restrict__ wk,
    const float* __restrict__ wv,
    _Float16* __restrict__ Kp, _Float16* __restrict__ Vp)
{
    const int slot = blockIdx.y;
    const float* X  = kin + (size_t)slot * BS * D;
    const float* Wk = wk  + (size_t)slot * D * D;
    const float* Wv = wv  + (size_t)slot * D * D;
    const int b0 = blockIdx.x * 128;

    __shared__ _Float16 Al[128 * 40];
    __shared__ _Float16 Bk[256 * 40];
    __shared__ _Float16 Bv[256 * 40];

    const int tid = threadIdx.x;
    const int lane = tid & 31;
    const int w   = tid >> 5;
    const int wm  = w & 3;
    const int wn  = w >> 2;
    const int hl  = lane >> 4;
    const int l16 = lane & 15;

    v8f accK[2][4], accV[2][4];
#pragma unroll
    for (int mi = 0; mi < 2; ++mi)
#pragma unroll
        for (int ni = 0; ni < 4; ++ni) { accK[mi][ni] = (v8f){}; accV[mi][ni] = (v8f){}; }

    for (int kt = 0; kt < 8; ++kt) {
        const int k0 = kt * 32;
#pragma unroll
        for (int i = 0; i < 2; ++i) {
            int idx = tid + i * 512;
            int row = idx >> 3, c4 = idx & 7;
            f4 v = __builtin_nontemporal_load((const f4*)&X[(size_t)(b0 + row) * D + k0 + c4 * 4]);
            h4 hv = {(_Float16)v.x, (_Float16)v.y, (_Float16)v.z, (_Float16)v.w};
            *(h4*)&Al[row * 40 + c4 * 4] = hv;
        }
#pragma unroll
        for (int i = 0; i < 4; ++i) {
            int idx = tid + i * 512;
            int row = idx >> 6, c4 = idx & 63;
            f4 a = *(const f4*)&Wk[(size_t)(k0 + row) * D + c4 * 4];
            f4 b = *(const f4*)&Wv[(size_t)(k0 + row) * D + c4 * 4];
            Bk[(c4 * 4 + 0) * 40 + row] = (_Float16)a.x;
            Bk[(c4 * 4 + 1) * 40 + row] = (_Float16)a.y;
            Bk[(c4 * 4 + 2) * 40 + row] = (_Float16)a.z;
            Bk[(c4 * 4 + 3) * 40 + row] = (_Float16)a.w;
            Bv[(c4 * 4 + 0) * 40 + row] = (_Float16)b.x;
            Bv[(c4 * 4 + 1) * 40 + row] = (_Float16)b.y;
            Bv[(c4 * 4 + 2) * 40 + row] = (_Float16)b.z;
            Bv[(c4 * 4 + 3) * 40 + row] = (_Float16)b.w;
        }
        __syncthreads();
        v16h a0 = load_a_frag(Al, wm * 32 + l16,      40, hl);
        v16h a1 = load_a_frag(Al, wm * 32 + 16 + l16, 40, hl);
#pragma unroll
        for (int ni = 0; ni < 4; ++ni) {
            v16h bf = load_b_frag(Bk, wn * 64 + ni * 16 + l16, 40, hl);
            accK[0][ni] = __builtin_amdgcn_wmma_f32_16x16x32_f16(false, a0, false, bf, (short)0, accK[0][ni], false, false);
            accK[1][ni] = __builtin_amdgcn_wmma_f32_16x16x32_f16(false, a1, false, bf, (short)0, accK[1][ni], false, false);
            v16h bg = load_b_frag(Bv, wn * 64 + ni * 16 + l16, 40, hl);
            accV[0][ni] = __builtin_amdgcn_wmma_f32_16x16x32_f16(false, a0, false, bg, (short)0, accV[0][ni], false, false);
            accV[1][ni] = __builtin_amdgcn_wmma_f32_16x16x32_f16(false, a1, false, bg, (short)0, accV[1][ni], false, false);
        }
        __syncthreads();
    }
#pragma unroll
    for (int mi = 0; mi < 2; ++mi)
#pragma unroll
        for (int ni = 0; ni < 4; ++ni)
#pragma unroll
            for (int r = 0; r < 8; ++r) {
                int brow = b0 + wm * 32 + mi * 16 + r + hl * 8;
                int col  = wn * 64 + ni * 16 + l16;
                size_t off = (size_t)brow * (NS * D) + slot * D + col;
                Kp[off] = (_Float16)accK[mi][ni][r];
                Vp[off] = (_Float16)accV[mi][ni][r];
            }
}

// -------------------- Phase 2: fused attention per batch element --------------------
// TDM 2D copy: 32 rows x 256 f16, row stride 256, into LDS with 8-half row padding
// (pad_interval = 128 dwords -> code 6, pad_amount = 4 dwords -> code 3) => LDS stride 264.
__device__ __forceinline__ void tdm_load_tile(const _Float16* g, unsigned lds_off) {
    unsigned lo = (unsigned)(size_t)g;
    unsigned hi = (unsigned)((size_t)g >> 32);
    v4u g0 = {1u,                                   // count=1, user descriptor
              lds_off,                              // lds_addr
              lo,                                   // global_addr[31:0]
              (hi & 0x01FFFFFFu) | 0x80000000u};    // global_addr[56:32] | type=2
    v8i g1 = {(int)((1u << 16) | (1u << 20) | (6u << 22) | (3u << 25)), // data_size=2B, pad en/intvl/amt
              (int)(256u << 16),                    // tensor_dim0 = 256
              (int)(32u  << 16),                    // tensor_dim1 = 32
              (int)(256u << 16),                    // tile_dim0 = 256
              32,                                   // tile_dim1 = 32
              256, 0, 0};                           // tensor_dim0_stride = 256
    v4i gz4 = {0, 0, 0, 0};
    v8i gz8 = {0, 0, 0, 0, 0, 0, 0, 0};
    __builtin_amdgcn_tensor_load_to_lds(g0, g1, gz4, gz4, gz8, 0);
}

__global__ __launch_bounds__(128) void attn_kernel(
    const _Float16* __restrict__ Qp, const _Float16* __restrict__ Kp,
    const _Float16* __restrict__ Vp, float* __restrict__ out)
{
    const int b = blockIdx.x;
    __shared__ __align__(512) _Float16 Qs[32 * 264];
    __shared__ __align__(512) _Float16 Ks[32 * 264];
    __shared__ _Float16 Vt[256 * 40];   // V transposed [o][m]
    __shared__ float    Lg[32 * 33];
    __shared__ _Float16 Ps[32 * 40];

    const int tid  = threadIdx.x;
    const int lane = tid & 31;
    const int w    = tid >> 5;
    const int hl   = lane >> 4;
    const int l16  = lane & 15;
    const size_t base = (size_t)b * (NS * D);

    if (w == 0) {                       // one wave issues both DMA descriptors
        tdm_load_tile(&Qp[base], (unsigned)(size_t)&Qs[0]);
        tdm_load_tile(&Kp[base], (unsigned)(size_t)&Ks[0]);
    }
    // V staged manually (TDM cannot transpose)
#pragma unroll
    for (int i = 0; i < 8; ++i) {
        int idx = tid + i * 128;
        int n = idx >> 5, c = idx & 31;
        h8 vv = __builtin_nontemporal_load((const h8*)&Vp[base + n * D + c * 8]);
#pragma unroll
        for (int e = 0; e < 8; ++e) Vt[(c * 8 + e) * 40 + n] = vv[e];
    }
    __builtin_amdgcn_s_wait_tensorcnt((short)0);   // no-op for waves 1..3
    __syncthreads();

    // logits = Q'K'^T / 16 : 2x2 WMMA tiles, K=256
    {
        const int wm = w & 1, wn = w >> 1;
        const int row = wm * 16 + l16;
        const int key = wn * 16 + l16;
        v8f c = (v8f){};
        for (int kt = 0; kt < 8; ++kt) {
            v16h a, bb;
#pragma unroll
            for (int v = 0; v < 8; ++v) {
                int ka = kt * 32 + (v & 3) * 2 + (v >> 2) * 16 + hl * 8;
                a[2 * v]     = Qs[row * 264 + ka];
                a[2 * v + 1] = Qs[row * 264 + ka + 1];
                int kb = kt * 32 + (v & 3) * 2 + (v >> 2) * 16 + hl * 8;
                bb[2 * v]     = Ks[key * 264 + kb];
                bb[2 * v + 1] = Ks[key * 264 + kb + 1];
            }
            c = __builtin_amdgcn_wmma_f32_16x16x32_f16(false, a, false, bb, (short)0, c, false, false);
        }
#pragma unroll
        for (int r = 0; r < 8; ++r)
            Lg[(wm * 16 + r + hl * 8) * 33 + wn * 16 + l16] = c[r] * 0.0625f;
    }
    __syncthreads();

    // softmax, one 32-wide row per lane of wave 0
    if (tid < 32) {
        float m = -1e30f;
#pragma unroll
        for (int j = 0; j < 32; ++j) m = fmaxf(m, Lg[tid * 33 + j]);
        float e[32];
        float s = 0.f;
#pragma unroll
        for (int j = 0; j < 32; ++j) { e[j] = __expf(Lg[tid * 33 + j] - m); s += e[j]; }
        float inv = 1.f / s;
#pragma unroll
        for (int j = 0; j < 32; ++j) Ps[tid * 40 + j] = (_Float16)(e[j] * inv);
    }
    __syncthreads();

    // out = P (32x32) @ V (32x256): each wave: 16 rows x 128 cols, K=32
    {
        const int mi = w & 1;
        const int ob = (w >> 1) * 128;
        v16h a = load_a_frag(Ps, mi * 16 + l16, 40, hl);
#pragma unroll
        for (int ni = 0; ni < 8; ++ni) {
            const int o = ob + ni * 16 + l16;
            v16h bb = load_b_frag(Vt, o, 40, hl);
            v8f c = (v8f){};
            c = __builtin_amdgcn_wmma_f32_16x16x32_f16(false, a, false, bb, (short)0, c, false, false);
#pragma unroll
            for (int r = 0; r < 8; ++r) {
                int n = mi * 16 + r + hl * 8;
                __builtin_nontemporal_store(c[r], &out[(size_t)n * (BS * D) + (size_t)b * D + o]);
            }
        }
    }
}

extern "C" void kernel_launch(void* const* d_in, const int* in_sizes, int n_in,
                              void* d_out, int out_size, void* d_ws, size_t ws_size,
                              hipStream_t stream) {
    const float* q  = (const float*)d_in[0];
    const float* k  = (const float*)d_in[1];
    const float* wq = (const float*)d_in[2];
    const float* wk = (const float*)d_in[3];
    const float* wv = (const float*)d_in[4];

    const size_t elems = (size_t)NS * BS * D;
    _Float16* Qp = (_Float16*)d_ws;
    _Float16* Kp = Qp + elems;
    _Float16* Vp = Kp + elems;

    dim3 g1(BS / 128, NS);
    proj_q_kernel <<<g1, 512, 0, stream>>>(q, wq, Qp);
    proj_kv_kernel<<<g1, 512, 0, stream>>>(k, wk, wv, Kp, Vp);
    attn_kernel   <<<BS, 128, 0, stream>>>(Qp, Kp, Vp, (float*)d_out);
}
